// KernalAnsatz_65481071401623
// MI455X (gfx1250) — matrix operand order repaired and checked by reading, not compiled, not run on hardware
//
#include <hip/hip_runtime.h>

typedef __attribute__((ext_vector_type(16))) _Float16 v16h;
typedef __attribute__((ext_vector_type(8)))  _Float16 v8h;
typedef __attribute__((ext_vector_type(8)))  float    v8f;

#define D   256
#define BM  128
#define BN  128
#define BK  32
#define LDS_STRIDE 40   // halves per LDS row (32 + 8 pad -> conflict-free b128 reads)

// ---------------- gamma: tiny MLP on x[0] ----------------
__global__ void rbf_gamma_kernel(const float* __restrict__ x,
                                 const float* __restrict__ W1,
                                 const float* __restrict__ b1,
                                 const float* __restrict__ W2,
                                 const float* __restrict__ b2,
                                 float* __restrict__ gamma_out) {
  __shared__ float h[16];
  int lane = threadIdx.x;
  if (lane < 16) {
    float acc = b1[lane];
    const float* w = W1 + lane * D;
    for (int k = 0; k < D; ++k) acc = fmaf(x[k], w[k], acc);
    h[lane] = fmaxf(acc, 0.0f);
  }
  __syncthreads();
  if (lane == 0) {
    float z = b2[0];
    for (int j = 0; j < 16; ++j) z = fmaf(h[j], W2[j], z);
    float sp = (z > 20.0f) ? z : log1pf(__expf(z));   // softplus
    gamma_out[0] = sp + 1e-6f;
  }
}

// ------- row prep: f32 -> f16 copy + squared row norms (one wave per row) -------
__global__ void __launch_bounds__(256)
rbf_rowprep_kernel(const float* __restrict__ src,
                   _Float16* __restrict__ dst,
                   float* __restrict__ sq,
                   int rows) {
  int gid  = blockIdx.x * blockDim.x + threadIdx.x;
  int row  = gid >> 5;
  int lane = threadIdx.x & 31;
  if (row >= rows) return;

  const float4* s = (const float4*)(src + (size_t)row * D);
  float4 a = s[lane * 2 + 0];
  float4 b = s[lane * 2 + 1];
  float ss = a.x*a.x + a.y*a.y + a.z*a.z + a.w*a.w
           + b.x*b.x + b.y*b.y + b.z*b.z + b.w*b.w;
#pragma unroll
  for (int o = 16; o > 0; o >>= 1) ss += __shfl_xor(ss, o, 32);

  v8h hv;
  hv[0] = (_Float16)a.x; hv[1] = (_Float16)a.y;
  hv[2] = (_Float16)a.z; hv[3] = (_Float16)a.w;
  hv[4] = (_Float16)b.x; hv[5] = (_Float16)b.y;
  hv[6] = (_Float16)b.z; hv[7] = (_Float16)b.w;
  ((v8h*)(dst + (size_t)row * D))[lane] = hv;

  if (lane == 0) sq[row] = ss;
}

// ---------------- main WMMA GEMM + RBF epilogue ----------------
__global__ void __launch_bounds__(256)
rbf_gemm_kernel(const _Float16* __restrict__ xh, const _Float16* __restrict__ yh,
                const float* __restrict__ xsq,  const float* __restrict__ ysq,
                const float* __restrict__ gamma_p, float* __restrict__ out,
                int n, int m) {
  __shared__ _Float16 xs[BM * LDS_STRIDE];
  __shared__ _Float16 ys[BN * LDS_STRIDE];

  const int tid  = threadIdx.x;
  const int lane = tid & 31;
  const int wave = tid >> 5;
  const int wm   = wave >> 1;   // 0..3 : 32-row strip of M
  const int wn   = wave & 1;    // 0..1 : 64-col strip of N

  const int row0 = blockIdx.y * BM;
  const int col0 = blockIdx.x * BN;

  v8f acc[2][4] = {};

  // per-lane fragment geometry (16-bit WMMA layouts, wave32)
  const int fr  = lane & 15;          // M (A) or N (B) index within 16
  const int akb = (lane >> 4) * 8;    // A: K blocks of 8, second block at +16
  const int bkb = (lane >> 4) * 8;    // B: K blocks of 8, second block at +16

  // tile staging: thread t loads 16 halves (one half-row of a 32-wide tile)
  const int ldr = tid >> 1;           // 0..127
  const int ldc = (tid & 1) * 16;     // 0 or 16

  union AF { v16h v; v8h h[2]; };

  for (int k0 = 0; k0 < D; k0 += BK) {
    {
      const v8h* g = (const v8h*)(xh + (size_t)(row0 + ldr) * D + k0 + ldc);
      v8h* l = (v8h*)(xs + ldr * LDS_STRIDE + ldc);
      l[0] = g[0];
      l[1] = g[1];
    }
    {
      const v8h* g = (const v8h*)(yh + (size_t)(col0 + ldr) * D + k0 + ldc);
      v8h* l = (v8h*)(ys + ldr * LDS_STRIDE + ldc);
      l[0] = g[0];
      l[1] = g[1];
    }
    __syncthreads();

    AF a[2], b[4];
#pragma unroll
    for (int i = 0; i < 2; ++i) {
      const _Float16* r = xs + (wm * 32 + i * 16 + fr) * LDS_STRIDE;
      a[i].h[0] = *(const v8h*)(r + akb);
      a[i].h[1] = *(const v8h*)(r + akb + 16);
    }
#pragma unroll
    for (int j = 0; j < 4; ++j) {
      const _Float16* r = ys + (wn * 64 + j * 16 + fr) * LDS_STRIDE;
      b[j].h[0] = *(const v8h*)(r + bkb);
      b[j].h[1] = *(const v8h*)(r + bkb + 16);
    }

#pragma unroll
    for (int i = 0; i < 2; ++i)
#pragma unroll
      for (int j = 0; j < 4; ++j)
        acc[i][j] = __builtin_amdgcn_wmma_f32_16x16x32_f16(
            false, a[i].v, false, b[j].v, (short)0, acc[i][j], false, false);

    __syncthreads();
  }

  // epilogue: dist^2 = |x|^2 + |y|^2 - 2*cross ; out = exp(-gamma * dist^2)
  const float gamma = gamma_p[0];
  const int mb = (lane >> 4) * 8;     // C/D layout: lanes 16-31 hold M+8
#pragma unroll
  for (int i = 0; i < 2; ++i) {
    const int rbase = row0 + wm * 32 + i * 16 + mb;
    float xv[8];
#pragma unroll
    for (int v = 0; v < 8; ++v) xv[v] = xsq[rbase + v];
#pragma unroll
    for (int j = 0; j < 4; ++j) {
      const int col = col0 + wn * 64 + j * 16 + fr;
      const float yv = ysq[col];
#pragma unroll
      for (int v = 0; v < 8; ++v) {
        float d2 = fmaxf(xv[v] + yv - 2.0f * acc[i][j][v], 0.0f);
        out[(size_t)(rbase + v) * m + col] = __expf(-gamma * d2);
      }
    }
  }
}

extern "C" void kernel_launch(void* const* d_in, const int* in_sizes, int n_in,
                              void* d_out, int out_size, void* d_ws, size_t ws_size,
                              hipStream_t stream) {
  const float* x  = (const float*)d_in[0];
  const float* y  = (const float*)d_in[1];
  const float* W1 = (const float*)d_in[2];
  const float* b1 = (const float*)d_in[3];
  const float* W2 = (const float*)d_in[4];
  const float* b2 = (const float*)d_in[5];

  const int n = in_sizes[0] / D;   // 8192
  const int m = in_sizes[1] / D;   // 8192

  // workspace layout: [gamma pad 256B][xh f16 n*D][yh f16 m*D][xsq n][ysq m]
  char*     ws    = (char*)d_ws;
  float*    gamma = (float*)ws;
  _Float16* xh    = (_Float16*)(ws + 256);
  _Float16* yh    = xh + (size_t)n * D;
  float*    xsq   = (float*)(yh + (size_t)m * D);
  float*    ysq   = xsq + n;
  float*    outf  = (float*)d_out;

  rbf_gamma_kernel<<<1, 32, 0, stream>>>(x, W1, b1, W2, b2, gamma);
  rbf_rowprep_kernel<<<(n + 7) / 8, 256, 0, stream>>>(x, xh, xsq, n);
  rbf_rowprep_kernel<<<(m + 7) / 8, 256, 0, stream>>>(y, yh, ysq, m);

  dim3 grid(m / BN, n / BM);
  rbf_gemm_kernel<<<grid, 256, 0, stream>>>(xh, yh, xsq, ysq, gamma, outf, n, m);
}
